// RNNBlock_82145544503296
// MI455X (gfx1250) — matrix operand order repaired and checked by reading, not compiled, not run on hardware
//
#include <hip/hip_runtime.h>

// Problem constants (match reference)
#define BDIM 8
#define LDIM 8192
#define DDIM 256
#define HDIM 256
#define MTOT (BDIM * LDIM)      // 65536 rows total
#define TCH  128                // scan chunk length
#define NCH  (LDIM / TCH)       // 64 chunks per sequence

typedef __attribute__((ext_vector_type(16))) __bf16 v16bf;
typedef __attribute__((ext_vector_type(8)))  float  v8f;
typedef __attribute__((ext_vector_type(4)))  int    v4i;

// ---------------------------------------------------------------------------
// gfx1250 async global->LDS copy (ASYNCcnt-tracked), with sync fallback.
// Builtin signature (from hipcc diagnostic): (v4i AS1*, v4i AS3*, imm, imm)
// ---------------------------------------------------------------------------
#if defined(__has_builtin)
#if __has_builtin(__builtin_amdgcn_global_load_async_to_lds_b128) && \
    __has_builtin(__builtin_amdgcn_s_wait_asynccnt)
#define USE_ASYNC_LDS 1
#endif
#endif

#if defined(USE_ASYNC_LDS)
typedef __attribute__((address_space(1))) v4i as1_v4i;
typedef __attribute__((address_space(3))) v4i as3_v4i;
__device__ __forceinline__ void async_copy16(const void* gsrc, void* ldst) {
    // generic->AS1: value-identical; generic->AS3: LDS offset lives in low 32 bits
    __builtin_amdgcn_global_load_async_to_lds_b128(
        (as1_v4i*)(uintptr_t)gsrc,
        (as3_v4i*)(uint32_t)(uintptr_t)ldst,
        0, 0);
}
#endif

// ---------------------------------------------------------------------------
// Prep: cast weights to bf16 layouts the GEMMs want, build gamma column scale.
//   Wt_bu : 512x256  rows 0..255 = B_re, 256..511 = B_im
//   Wt_y  : 256x512  row o = [C_re[o,:], -C_im[o,:]]
//   Wt_w  : 256x256  = W (out[m,d] = sum_o g[m,o] * W[d,o])
//   gammaN: 512      gamma[h] replicated for re/im columns
// ---------------------------------------------------------------------------
__global__ void prep_kernel(const float* __restrict__ Br, const float* __restrict__ Bi,
                            const float* __restrict__ Cr, const float* __restrict__ Ci,
                            const float* __restrict__ W,  const float* __restrict__ nu_log,
                            __bf16* __restrict__ Wt_bu, __bf16* __restrict__ Wt_y,
                            __bf16* __restrict__ Wt_w,  float* __restrict__ gammaN) {
    int i = blockIdx.x * 256 + threadIdx.x;
    if (i < 131072) {                    // Wt_bu
        int n = i >> 8, k = i & 255;
        float v = (n < 256) ? Br[n * 256 + k] : Bi[(n - 256) * 256 + k];
        Wt_bu[i] = (__bf16)v;
    } else if (i < 262144) {             // Wt_y
        int j = i - 131072;
        int o = j >> 9, k = j & 511;
        float v = (k < 256) ? Cr[o * 256 + k] : -Ci[o * 256 + (k - 256)];
        Wt_y[j] = (__bf16)v;
    } else if (i < 327680) {             // Wt_w
        int j = i - 262144;
        Wt_w[j] = (__bf16)W[j];
    } else if (i < 328192) {             // gammaN
        int n = i - 327680;
        float nu = expf(nu_log[n & 255]);
        gammaN[n] = sqrtf(fmaxf(0.f, 1.f - expf(-2.f * nu)));
    }
}

// ---------------------------------------------------------------------------
// LayerNorm: one wave (32 lanes) per row of 256, 8 rows per 256-thread block.
// ---------------------------------------------------------------------------
__global__ void ln_kernel(const float* __restrict__ x, const float* __restrict__ sc,
                          const float* __restrict__ bi, __bf16* __restrict__ xn) {
    const int wid = threadIdx.x >> 5, lane = threadIdx.x & 31;
    const int row = blockIdx.x * 8 + wid;
    const float* xr = x + (size_t)row * DDIM;
    float v[8];
    float s = 0.f, s2 = 0.f;
#pragma unroll
    for (int i = 0; i < 8; ++i) {
        v[i] = xr[lane + 32 * i];
        s += v[i];
        s2 += v[i] * v[i];
    }
#pragma unroll
    for (int m = 16; m >= 1; m >>= 1) {
        s  += __shfl_xor(s, m, 32);
        s2 += __shfl_xor(s2, m, 32);
    }
    const float mu   = s * (1.f / DDIM);
    const float var  = s2 * (1.f / DDIM) - mu * mu;
    const float rstd = rsqrtf(var + 1e-6f);
    __bf16* xo = xn + (size_t)row * DDIM;
#pragma unroll
    for (int i = 0; i < 8; ++i) {
        int d = lane + 32 * i;
        xo[d] = (__bf16)((v[i] - mu) * rstd * sc[d] + bi[d]);
    }
}

// ---------------------------------------------------------------------------
// bf16 GEMM, fp32 WMMA accumulate: out[m,n] = sum_k A[m,k]*Wt[n,k]
// Block = 8 waves, all sharing one 64-column weight slab staged in LDS via
// async global->LDS (gfx1250) then read as B fragments with ds_load_b128.
// Wave tile: 16(M) x 64(N); A fragments stream from global, pipelined 1 deep.
// Fragment layouts per CDNA5 ISA tables (wave32):
//   A (16x32 bf16): lane(g=lane>>4, m=lane&15) holds K runs [k0+8g,+8) and
//                   [k0+16+8g,+8)
//   B (32x16 bf16): lane(g, n=lane&15) holds K runs [k0+8g,+8) and [k0+16+8g,+8) of column n
//   C/D (16x16 f32): element r of v8f is (m = m_base+8g+r, n = n_base+(lane&15))
// MODE 0: Bu GEMM, scale column by gammaN          -> fp32 out (M x 512)
// MODE 1: y GEMM, + D_skip*xn, gelu(tanh)          -> bf16 out (M x 256)
// MODE 2: final GEMM, + bias + residual x          -> fp32 out (d_out)
// ---------------------------------------------------------------------------
template <int MODE, int K, int N>
__global__ void gemm_kernel(const __bf16* __restrict__ A, const __bf16* __restrict__ Wt,
                            const float* __restrict__ p0,   // gamma / D_skip / bias
                            const float* __restrict__ p1,   // MODE2: residual x (fp32)
                            const __bf16* __restrict__ pbf, // MODE1: xn (bf16)
                            float* __restrict__ outF, __bf16* __restrict__ outB) {
    __shared__ __align__(16) __bf16 wsh[64 * K];

    const int wid = threadIdx.x >> 5, lane = threadIdx.x & 31;
    const int g = lane >> 4, ln = lane & 15;
    // 512 blocks per n-slab (MTOT/16/8); all 8 waves of a block share n_base
    const int nt = blockIdx.x >> 9;
    const int m_base = ((blockIdx.x & 511) * 8 + wid) * 16;
    const int n_base = nt * 64;

    // ---- stage the contiguous 64-row weight slab into LDS ----
    const __bf16* slab = Wt + (size_t)n_base * K;        // 64*K contiguous bf16
    constexpr int CHUNKS = (64 * K) / 8 / 256;           // b128 chunks per thread
#pragma unroll
    for (int i = 0; i < CHUNKS; ++i) {
        const int c = threadIdx.x + 256 * i;             // 16B chunk index
#if defined(USE_ASYNC_LDS)
        async_copy16(slab + c * 8, wsh + c * 8);
#else
        *(uint4*)(wsh + c * 8) = *(const uint4*)(slab + c * 8);
#endif
    }
#if defined(USE_ASYNC_LDS)
    __builtin_amdgcn_s_wait_asynccnt(0);
#endif
    __syncthreads();

    v8f acc[4] = {};

    union uA { v16bf v; uint4 u[2]; };
    const __bf16* arow = A + (size_t)(m_base + ln) * K + 8 * g;
    uA cur, nxt;
    cur.u[0] = *(const uint4*)(arow);
    cur.u[1] = *(const uint4*)(arow + 16);
#pragma unroll
    for (int k0 = 0; k0 < K; k0 += 32) {
        if (k0 + 32 < K) {                               // pipeline next A frag
            nxt.u[0] = *(const uint4*)(arow + k0 + 32);
            nxt.u[1] = *(const uint4*)(arow + k0 + 48);
        }
#pragma unroll
        for (int j = 0; j < 4; ++j) {
            uA ub;
            const __bf16* wp = wsh + (j * 16 + ln) * K + k0 + 8 * g;
            ub.u[0] = *(const uint4*)(wp);
            ub.u[1] = *(const uint4*)(wp + 16);
            acc[j] = __builtin_amdgcn_wmma_f32_16x16x32_bf16(
                false, cur.v, false, ub.v, (short)0, acc[j], false, false);
        }
        if (k0 + 32 < K) cur = nxt;
    }

#pragma unroll
    for (int j = 0; j < 4; ++j) {
#pragma unroll
        for (int r = 0; r < 8; ++r) {
            const int m = m_base + 8 * g + r;
            const int n = n_base + j * 16 + ln;
            float v = acc[j][r];
            if (MODE == 0) {
                outF[(size_t)m * N + n] = v * p0[n];
            } else if (MODE == 1) {
                v += p0[n] * (float)pbf[(size_t)m * DDIM + n];
                const float u = 0.7978845608028654f * (v + 0.044715f * v * v * v);
                v = 0.5f * v * (1.f + tanhf(u));
                outB[(size_t)m * N + n] = (__bf16)v;
            } else {
                outF[(size_t)m * N + n] = v + p0[n] + p1[(size_t)m * DDIM + n];
            }
        }
    }
}

// ---------------------------------------------------------------------------
// Chunked diagonal complex scan: h_t = lam*h_{t-1} + Bu_t
// Bu layout: M x 512 fp32 (cols 0..255 = re, 256..511 = im); h emitted as
// bf16 M x 512 for the following GEMM. State kept in fp32 throughout.
// ---------------------------------------------------------------------------
__device__ __forceinline__ void lam_of(const float* nu_log, const float* th_log,
                                       int h, float& lr, float& li) {
    const float nu = expf(nu_log[h]);
    const float th = expf(th_log[h]);
    const float r = expf(-nu);
    lr = r * cosf(th);
    li = r * sinf(th);
}

__global__ void scan_carry_kernel(const float* __restrict__ Bu,
                                  const float* __restrict__ nu_log,
                                  const float* __restrict__ th_log,
                                  float2* __restrict__ carry) {
    const int tid = blockIdx.x * blockDim.x + threadIdx.x; // B*NCH*H threads
    const int h = tid & (HDIM - 1);
    const int idx = tid >> 8;
    const int chunk = idx & (NCH - 1);
    const int b = idx >> 6;
    float lr, li;
    lam_of(nu_log, th_log, h, lr, li);
    const size_t m0 = (size_t)b * LDIM + (size_t)chunk * TCH;
    float hr = 0.f, hi = 0.f;
    for (int t = 0; t < TCH; ++t) {
        const float* row = Bu + (m0 + t) * 512;
        const float br = row[h], bim = row[256 + h];
        const float nhr = lr * hr - li * hi + br;
        const float nhi = lr * hi + li * hr + bim;
        hr = nhr; hi = nhi;
    }
    carry[tid] = make_float2(hr, hi);
}

__global__ void scan_prefix_kernel(const float* __restrict__ nu_log,
                                   const float* __restrict__ th_log,
                                   float2* __restrict__ carry) {
    const int tid = blockIdx.x * blockDim.x + threadIdx.x; // B*H threads
    const int h = tid & (HDIM - 1);
    const int b = tid >> 8;
    const float nu = expf(nu_log[h]);
    const float th = expf(th_log[h]);
    const float rT = expf(-(float)TCH * nu);
    const float thT = (float)TCH * th;
    const float aTr = rT * cosf(thT), aTi = rT * sinf(thT);
    float Sr = 0.f, Si = 0.f;
    for (int k = 0; k < NCH; ++k) {
        const int ci = (b * NCH + k) * HDIM + h;
        const float2 c = carry[ci];
        carry[ci] = make_float2(Sr, Si);       // entering state for chunk k
        const float nSr = aTr * Sr - aTi * Si + c.x;
        const float nSi = aTr * Si + aTi * Sr + c.y;
        Sr = nSr; Si = nSi;
    }
}

__global__ void scan_apply_kernel(const float* __restrict__ Bu,
                                  const float* __restrict__ nu_log,
                                  const float* __restrict__ th_log,
                                  const float2* __restrict__ carry,
                                  __bf16* __restrict__ hbf) {
    const int tid = blockIdx.x * blockDim.x + threadIdx.x;
    const int h = tid & (HDIM - 1);
    const int idx = tid >> 8;
    const int chunk = idx & (NCH - 1);
    const int b = idx >> 6;
    float lr, li;
    lam_of(nu_log, th_log, h, lr, li);
    const size_t m0 = (size_t)b * LDIM + (size_t)chunk * TCH;
    const float2 S = carry[tid];
    float hr = S.x, hi = S.y;
    for (int t = 0; t < TCH; ++t) {
        const size_t m = m0 + t;
        const float* row = Bu + m * 512;
        const float br = row[h], bim = row[256 + h];
        const float nhr = lr * hr - li * hi + br;
        const float nhi = lr * hi + li * hr + bim;
        hr = nhr; hi = nhi;
        hbf[m * 512 + h]       = (__bf16)hr;
        hbf[m * 512 + 256 + h] = (__bf16)hi;
    }
}

// ---------------------------------------------------------------------------
// Launch
// ---------------------------------------------------------------------------
extern "C" void kernel_launch(void* const* d_in, const int* in_sizes, int n_in,
                              void* d_out, int out_size, void* d_ws, size_t ws_size,
                              hipStream_t stream) {
    const float* x        = (const float*)d_in[0];
    const float* ln_scale = (const float*)d_in[1];
    const float* ln_bias  = (const float*)d_in[2];
    const float* nu_log   = (const float*)d_in[3];
    const float* th_log   = (const float*)d_in[4];
    const float* B_re     = (const float*)d_in[5];
    const float* B_im     = (const float*)d_in[6];
    const float* C_re     = (const float*)d_in[7];
    const float* C_im     = (const float*)d_in[8];
    const float* D_skip   = (const float*)d_in[9];
    const float* Wmat     = (const float*)d_in[10];
    const float* bvec     = (const float*)d_in[11];

    char* ws = (char*)d_ws;
    size_t off = 0;
    auto alloc = [&](size_t bytes) -> void* {
        void* p = ws + off;
        off += (bytes + 255) & ~(size_t)255;
        return p;
    };
    __bf16* xn     = (__bf16*)alloc((size_t)MTOT * DDIM * 2);
    __bf16* Wt_bu  = (__bf16*)alloc((size_t)512 * 256 * 2);
    __bf16* Wt_y   = (__bf16*)alloc((size_t)256 * 512 * 2);
    __bf16* Wt_w   = (__bf16*)alloc((size_t)256 * 256 * 2);
    float*  gammaN = (float*) alloc((size_t)512 * 4);
    float*  Bu     = (float*) alloc((size_t)MTOT * 512 * 4);
    float2* carry  = (float2*)alloc((size_t)BDIM * NCH * HDIM * 8);
    __bf16* hbf    = (__bf16*)alloc((size_t)MTOT * 512 * 2);
    __bf16* gbf    = (__bf16*)alloc((size_t)MTOT * 256 * 2);

    // 1. weights -> bf16, gamma
    prep_kernel<<<(328192 + 255) / 256, 256, 0, stream>>>(
        B_re, B_im, C_re, C_im, Wmat, nu_log, Wt_bu, Wt_y, Wt_w, gammaN);

    // 2. layernorm
    ln_kernel<<<MTOT / 8, 256, 0, stream>>>(x, ln_scale, ln_bias, xn);

    // 3. Bu = gamma * (xn @ [B_re;B_im]^T)   (M x 512, fp32)
    gemm_kernel<0, 256, 512><<<512 * (512 / 64), 256, 0, stream>>>(
        xn, Wt_bu, gammaN, nullptr, nullptr, Bu, nullptr);

    // 4. chunked scan
    scan_carry_kernel<<<(BDIM * NCH * HDIM) / 256, 256, 0, stream>>>(Bu, nu_log, th_log, carry);
    scan_prefix_kernel<<<(BDIM * HDIM) / 256, 256, 0, stream>>>(nu_log, th_log, carry);
    scan_apply_kernel<<<(BDIM * NCH * HDIM) / 256, 256, 0, stream>>>(Bu, nu_log, th_log, carry, hbf);

    // 5. y = [h_re|h_im] @ [C_re|-C_im]^T + D_skip*xn ; gelu -> bf16
    gemm_kernel<1, 512, 256><<<512 * (256 / 64), 256, 0, stream>>>(
        hbf, Wt_y, D_skip, nullptr, xn, nullptr, gbf);

    // 6. out = g @ W^T + b + x  (fp32)
    gemm_kernel<2, 256, 256><<<512 * (256 / 64), 256, 0, stream>>>(
        gbf, Wt_w, bvec, x, nullptr, (float*)d_out, nullptr);
}
